// AttentiveFP_20229295964906
// MI455X (gfx1250) — matrix operand order repaired and checked by reading, not compiled, not run on hardware
//
#include <hip/hip_runtime.h>
#include <hip/hip_bf16.h>
#include <math.h>

typedef __attribute__((ext_vector_type(16))) _Float16 v16h;
typedef __attribute__((ext_vector_type(8)))  _Float16 v8h;
typedef __attribute__((ext_vector_type(4)))  _Float16 v4h;
typedef __attribute__((ext_vector_type(8)))  float    v8f;

#define N_NODES 65536
#define N_EDGES 262144
#define IN_DIM  64
#define HD      256
#define EDD     16
#define NG      2048
#define NLAYERS 3
#define NSTEPS  3

// activation codes
#define ACT_NONE  0
#define ACT_LRELU 1
#define ACT_RELU  2
#define ACT_ELU   3

__device__ __forceinline__ float apply_act(float v, int act) {
    switch (act) {
        case ACT_LRELU: return v > 0.f ? v : 0.01f * v;
        case ACT_RELU:  return v > 0.f ? v : 0.f;
        case ACT_ELU:   return v > 0.f ? v : (expf(v) - 1.f);
        default:        return v;
    }
}

__device__ __forceinline__ unsigned fmap(float f) {
    unsigned u = __float_as_uint(f);
    return (u >> 31) ? ~u : (u | 0x80000000u);
}
__device__ __forceinline__ float funmap(unsigned u) {
    return __uint_as_float((u >> 31) ? (u & 0x7FFFFFFFu) : ~u);
}

__device__ __forceinline__ v4h cvt4(float4 v) {
    v4h r;
    r[0] = (_Float16)v.x; r[1] = (_Float16)v.y;
    r[2] = (_Float16)v.z; r[3] = (_Float16)v.w;
    return r;
}

// A fragment 16x32 f16: lanes 0-15 -> K {0..7,16..23}, lanes 16-31 -> K {8..15,24..31}
__device__ __forceinline__ v16h frag_a(const _Float16* row, int kb) {
    v8h lo = *(const v8h*)(row + kb);
    v8h hi = *(const v8h*)(row + 16 + kb);
    v16h r;
    #pragma unroll
    for (int i = 0; i < 8; i++) { r[i] = lo[i]; r[8 + i] = hi[i]; }
    return r;
}
// B fragment 32x16 f16: column = lane&15, lanes 0-15 -> K {0..7,16..23}, lanes 16-31 -> K {8..15,24..31} (kbb = 0|8)
__device__ __forceinline__ v16h frag_b(const _Float16* row, int kbb) {
    v8h lo = *(const v8h*)(row + kbb);
    v8h hi = *(const v8h*)(row + 16 + kbb);
    v16h r;
    #pragma unroll
    for (int i = 0; i < 8; i++) { r[i] = lo[i]; r[8 + i] = hi[i]; }
    return r;
}

// ---------------------------------------------------------------------------
// Tiled WMMA GEMM:  Y = act(A' @ W^T + bias)
//   GATHER=false: A' = A (leading dim K). Requires M % 128 == 0, K % 32 == 0
//                 (true for every call site) -> guard-free A staging/epilogue.
//   GATHER=true : A' row r = concat(A[rowIdx[r]] (K1 cols), A2[r] (K-K1 cols)),
//                 zero-padded to the next multiple of 32 (K=272 case).
//   W is [Nout, K] row-major.
//   Block tile 128x64, 8 waves (4 along M x 2 along N), wave tile 32x32,
//   K stepped by 32 using v_wmma_f32_16x16x32_f16 (f16 inputs, f32 accum).
//   All K / leading dims are multiples of 4 -> float4 staging, v4h LDS stores.
// ---------------------------------------------------------------------------
template <bool GATHER>
__global__ __launch_bounds__(256)
void gemm_kernel(const float* __restrict__ A, const int* __restrict__ rowIdx,
                 const float* __restrict__ A2, int K1,
                 const float* __restrict__ W, const float* __restrict__ bias,
                 float* __restrict__ Y, int M, int Nout, int K, int act)
{
    __shared__ _Float16 As[128][40];  // 128 rows x 32 K (+8 pad, keeps 16B align)
    __shared__ _Float16 Bs[64][40];   // 64 cols  x 32 K

    const int tid  = threadIdx.x;
    const int lane = tid & 31;
    const int wave = tid >> 5;
    const int rowBase = blockIdx.x * 128;
    const int colBase = blockIdx.y * 64;
    const int wm = (wave & 3) * 32;   // wave row offset within block tile
    const int wn = (wave >> 2) * 32;  // wave col offset within block tile
    const int K2 = K - K1;

    v8f c00 = {}, c01 = {}, c10 = {}, c11 = {};

    for (int k0 = 0; k0 < K; k0 += 32) {
        // stage A tile (f32 -> f16), vec4 granularity: 128*8 groups / 256 thr
        #pragma unroll
        for (int it = 0; it < 4; it++) {
            int i = tid + it * 256;
            int r  = i >> 3;
            int c4 = (i & 7) * 4;
            int row = rowBase + r;
            int k = k0 + c4;
            float4 v;
            if (GATHER) {
                v = make_float4(0.f, 0.f, 0.f, 0.f);
                if (k < K1) {
                    v = *(const float4*)(A + (size_t)rowIdx[row] * K1 + k);
                } else if (k < K) {
                    v = *(const float4*)(A2 + (size_t)row * K2 + (k - K1));
                }
            } else {
                v = *(const float4*)(A + (size_t)row * K + k);
            }
            *(v4h*)&As[r][c4] = cvt4(v);
        }
        // stage B tile: Bs[n][k] = W[colBase+n][k0+k]; 64*8 groups / 256 thr
        #pragma unroll
        for (int it = 0; it < 2; it++) {
            int i = tid + it * 256;
            int n  = i >> 3;
            int c4 = (i & 7) * 4;
            int col = colBase + n;
            int k = k0 + c4;
            float4 v = make_float4(0.f, 0.f, 0.f, 0.f);
            if (GATHER) {
                if (col < Nout && k < K)
                    v = *(const float4*)(W + (size_t)col * K + k);
            } else {
                if (col < Nout)
                    v = *(const float4*)(W + (size_t)col * K + k);
            }
            *(v4h*)&Bs[n][c4] = cvt4(v);
        }
        __syncthreads();

        const int mloc = lane & 15;
        const int kb   = (lane < 16) ? 0 : 8;
        const int kbb  = (lane < 16) ? 0 : 8;

        v16h a0 = frag_a(&As[wm + mloc][0], kb);
        v16h a1 = frag_a(&As[wm + 16 + mloc][0], kb);
        v16h b0 = frag_b(&Bs[wn + mloc][0], kbb);
        v16h b1 = frag_b(&Bs[wn + 16 + mloc][0], kbb);

        c00 = __builtin_amdgcn_wmma_f32_16x16x32_f16(false, a0, false, b0, (short)0, c00, false, false);
        c01 = __builtin_amdgcn_wmma_f32_16x16x32_f16(false, a0, false, b1, (short)0, c01, false, false);
        c10 = __builtin_amdgcn_wmma_f32_16x16x32_f16(false, a1, false, b0, (short)0, c10, false, false);
        c11 = __builtin_amdgcn_wmma_f32_16x16x32_f16(false, a1, false, b1, (short)0, c11, false, false);
        __syncthreads();
    }

    // D layout: VGPR i -> (lanes 0-15: M=i, N=lane) / (lanes 16-31: M=8+i, N=lane-16)
    // M % 128 == 0 at every call site -> no row guards.
    const int nLoc = lane & 15;
    const int mAdd = (lane < 16) ? 0 : 8;
    const int col0 = colBase + wn + nLoc;
    const int col1 = col0 + 16;
    const bool ok0 = col0 < Nout;
    const bool ok1 = col1 < Nout;
    const float bb0 = (bias && ok0) ? bias[col0] : 0.f;
    const float bb1 = (bias && ok1) ? bias[col1] : 0.f;
    #pragma unroll
    for (int i = 0; i < 8; i++) {
        size_t rowA = (size_t)(rowBase + wm + mAdd + i);
        size_t rowB = rowA + 16;
        if (ok0) {
            Y[rowA * Nout + col0] = apply_act(c00[i] + bb0, act);
            Y[rowB * Nout + col0] = apply_act(c10[i] + bb0, act);
        }
        if (ok1) {
            Y[rowA * Nout + col1] = apply_act(c01[i] + bb1, act);
            Y[rowB * Nout + col1] = apply_act(c11[i] + bb1, act);
        }
    }
}

// ---------------------------------------------------------------------------
// wave-per-row dot:  out[r] = dot(X[r, 0:256], w)
// ---------------------------------------------------------------------------
__global__ __launch_bounds__(256)
void rowdot_kernel(const float* __restrict__ X, const float* __restrict__ w,
                   float* __restrict__ out, int M)
{
    int r = blockIdx.x * 8 + (threadIdx.x >> 5);
    int lane = threadIdx.x & 31;
    if (r >= M) return;
    const float* row = X + (size_t)r * HD;
    float s = 0.f;
    #pragma unroll
    for (int c = 0; c < HD; c += 32) s += row[c + lane] * w[c + lane];
    for (int off = 16; off > 0; off >>= 1) s += __shfl_down(s, off, 32);
    if (lane == 0) out[r] = s;
}

// GATEConv edge score: eV[e] = lrelu( dot(xj[e], att_l) + grD[dst[e]] )
__global__ __launch_bounds__(256)
void edge_score_gate_kernel(const float* __restrict__ xj, const float* __restrict__ attl,
                            const float* __restrict__ grD, const int* __restrict__ dstIdx,
                            float* __restrict__ eV, int E)
{
    int e = blockIdx.x * 8 + (threadIdx.x >> 5);
    int lane = threadIdx.x & 31;
    if (e >= E) return;
    const float* row = xj + (size_t)e * HD;
    float s = 0.f;
    #pragma unroll
    for (int c = 0; c < HD; c += 32) s += row[c + lane] * attl[c + lane];
    for (int off = 16; off > 0; off >>= 1) s += __shfl_down(s, off, 32);
    if (lane == 0) {
        float v = s + grD[dstIdx[e]];
        eV[e] = v > 0.f ? v : 0.01f * v;
    }
}

// GATv2 edge score: eV[e] = sum_c lrelu(hl[src][c] + hr[dst][c]) * att[c]
__global__ __launch_bounds__(256)
void edge_score_v2_kernel(const float* __restrict__ hl, const float* __restrict__ hr,
                          const int* __restrict__ srcIdx, const int* __restrict__ dstIdx,
                          const float* __restrict__ att, float* __restrict__ eV, int E)
{
    int e = blockIdx.x * 8 + (threadIdx.x >> 5);
    int lane = threadIdx.x & 31;
    if (e >= E) return;
    int s_ = srcIdx ? srcIdx[e] : e;
    int d_ = dstIdx[e];
    const float* rl = hl + (size_t)s_ * HD;
    const float* rr = hr + (size_t)d_ * HD;
    float s = 0.f;
    #pragma unroll
    for (int c = 0; c < HD; c += 32) {
        float v = rl[c + lane] + rr[c + lane];
        v = v > 0.f ? v : 0.01f * v;
        s += v * att[c + lane];
    }
    for (int off = 16; off > 0; off >>= 1) s += __shfl_down(s, off, 32);
    if (lane == 0) eV[e] = s;
}

__global__ void seg_max_kernel(const float* __restrict__ eV, const int* __restrict__ dstIdx,
                               unsigned* __restrict__ segmax, int E)
{
    int e = blockIdx.x * 256 + threadIdx.x;
    if (e >= E) return;
    atomicMax(&segmax[dstIdx[e]], fmap(eV[e]));
}

__global__ void seg_expsum_kernel(float* __restrict__ eV, const int* __restrict__ dstIdx,
                                  const unsigned* __restrict__ segmax,
                                  float* __restrict__ segsum, int E)
{
    int e = blockIdx.x * 256 + threadIdx.x;
    if (e >= E) return;
    int d_ = dstIdx[e];
    float a = expf(eV[e] - funmap(segmax[d_]));
    eV[e] = a;
    atomicAdd(&segsum[d_], a);
}

// out[dst[e]] += Msrc[srcIdx?srcIdx[e]:e] * (alpha ? alpha[e]/segsum[dst[e]] : 1)
__global__ __launch_bounds__(256)
void scatter_kernel(const float* __restrict__ Msrc, const int* __restrict__ srcIdx,
                    const float* __restrict__ alpha, const float* __restrict__ segsum,
                    const int* __restrict__ dstIdx, float* __restrict__ out, int E)
{
    int e = blockIdx.x;
    int c = threadIdx.x;
    if (e >= E) return;
    int s_ = srcIdx ? srcIdx[e] : e;
    int d_ = dstIdx[e];
    float sc = 1.f;
    if (alpha) sc = alpha[e] / segsum[d_];
    atomicAdd(&out[(size_t)d_ * HD + c], Msrc[(size_t)s_ * HD + c] * sc);
}

__global__ void bias_act_kernel(float* __restrict__ X, const float* __restrict__ bias,
                                long total, int Nc, int act)
{
    long i = (long)blockIdx.x * blockDim.x + threadIdx.x;
    if (i >= total) return;
    int c = (int)(i % Nc);
    float v = X[i] + (bias ? bias[c] : 0.f);
    X[i] = apply_act(v, act);
}

// GRUCell combine; gi/gh already contain biases from the GEMMs.
__global__ __launch_bounds__(256)
void gru_kernel(const float* __restrict__ gi, const float* __restrict__ gh,
                const float* __restrict__ hprev, float* __restrict__ out,
                int M, int doRelu)
{
    int n = blockIdx.x;
    int c = threadIdx.x;
    if (n >= M) return;
    const float* gir = gi + (size_t)n * (3 * HD);
    const float* ghr = gh + (size_t)n * (3 * HD);
    float r = 1.f / (1.f + expf(-(gir[c] + ghr[c])));
    float z = 1.f / (1.f + expf(-(gir[HD + c] + ghr[HD + c])));
    float nn = tanhf(gir[2 * HD + c] + r * ghr[2 * HD + c]);
    float h = hprev[(size_t)n * HD + c];
    float o = (1.f - z) * nn + z * h;
    if (doRelu) o = o > 0.f ? o : 0.f;
    out[(size_t)n * HD + c] = o;
}

__global__ void fillf_kernel(float* p, float v, long n) {
    long i = (long)blockIdx.x * blockDim.x + threadIdx.x;
    if (i < n) p[i] = v;
}
__global__ void fillu_kernel(unsigned* p, unsigned v, long n) {
    long i = (long)blockIdx.x * blockDim.x + threadIdx.x;
    if (i < n) p[i] = v;
}

// ------------------------------ host side ----------------------------------
static inline void gemm(hipStream_t st, const float* A, const int* rowIdx, const float* A2,
                        int K1, const float* W, const float* bias, float* Y,
                        int M, int Nout, int K, int act)
{
    dim3 g((M + 127) / 128, (Nout + 63) / 64);
    if (rowIdx)
        gemm_kernel<true><<<g, 256, 0, st>>>(A, rowIdx, A2, K1, W, bias, Y, M, Nout, K, act);
    else
        gemm_kernel<false><<<g, 256, 0, st>>>(A, nullptr, nullptr, K, W, bias, Y, M, Nout, K, act);
}
static inline void fillf(hipStream_t st, float* p, float v, long n) {
    fillf_kernel<<<(unsigned)((n + 255) / 256), 256, 0, st>>>(p, v, n);
}
static inline void fillu(hipStream_t st, unsigned* p, unsigned v, long n) {
    fillu_kernel<<<(unsigned)((n + 255) / 256), 256, 0, st>>>(p, v, n);
}

extern "C" void kernel_launch(void* const* d_in, const int* in_sizes, int n_in,
                              void* d_out, int out_size, void* d_ws, size_t ws_size,
                              hipStream_t stream)
{
    const float* x0        = (const float*)d_in[0];
    const float* edge_attr = (const float*)d_in[1];
    const int*   eidx      = (const int*)d_in[2];
    const int*   batch     = (const int*)d_in[3];
    const float* lin1_w    = (const float*)d_in[4];
    const float* lin1_b    = (const float*)d_in[5];
    const float* g_lin1_w  = (const float*)d_in[6];
    const float* g_att_l   = (const float*)d_in[7];
    const float* g_att_r   = (const float*)d_in[8];
    const float* g_lin2_w  = (const float*)d_in[9];
    const float* g_bias    = (const float*)d_in[10];
    const float* gru0_wih  = (const float*)d_in[11];
    const float* gru0_whh  = (const float*)d_in[12];
    const float* gru0_bih  = (const float*)d_in[13];
    const float* gru0_bhh  = (const float*)d_in[14];
    const float* atom_Wl   = (const float*)d_in[15];
    const float* atom_Wr   = (const float*)d_in[16];
    const float* atom_att  = (const float*)d_in[17];
    const float* atom_bias = (const float*)d_in[18];
    const float* agru_wih  = (const float*)d_in[19];
    const float* agru_whh  = (const float*)d_in[20];
    const float* agru_bih  = (const float*)d_in[21];
    const float* agru_bhh  = (const float*)d_in[22];
    const float* mol_Wl    = (const float*)d_in[23];
    const float* mol_Wr    = (const float*)d_in[24];
    const float* mol_att   = (const float*)d_in[25];
    const float* mol_bias  = (const float*)d_in[26];
    const float* mgru_wih  = (const float*)d_in[27];
    const float* mgru_whh  = (const float*)d_in[28];
    const float* mgru_bih  = (const float*)d_in[29];
    const float* mgru_bhh  = (const float*)d_in[30];
    const float* mlp_w1    = (const float*)d_in[31];
    const float* mlp_b1    = (const float*)d_in[32];
    const float* mlp_w2    = (const float*)d_in[33];
    const float* mlp_b2    = (const float*)d_in[34];
    const float* mlp_w3    = (const float*)d_in[35];
    const float* mlp_b3    = (const float*)d_in[36];

    const int* src = eidx;
    const int* dst = eidx + N_EDGES;

    // workspace layout (floats)
    float* ws = (float*)d_ws;
    size_t o = 0;
    float* x1   = ws + o; o += (size_t)N_NODES * HD;      // node state
    float* hb   = ws + o; o += (size_t)N_NODES * HD;      // conv output h
    float* gi   = ws + o; o += (size_t)N_NODES * 3 * HD;  // GRU input gates
    float* gh   = ws + o; o += (size_t)N_NODES * 3 * HD;  // GRU hidden gates
    float* hl   = ws + o; o += (size_t)N_NODES * HD;
    float* hr   = ws + o; o += (size_t)N_NODES * HD;      // also reused for mol hr [G,HD]
    float* xj   = ws + o; o += (size_t)N_EDGES * HD;
    float* mm   = ws + o; o += (size_t)N_EDGES * HD;
    float* eV   = ws + o; o += (size_t)N_EDGES;           // edge scores / alpha (>=N for mol)
    float* grD  = ws + o; o += (size_t)N_NODES;
    unsigned* smax = (unsigned*)(ws + o); o += (size_t)N_NODES;
    float* ssum = ws + o; o += (size_t)N_NODES;
    float* outG = ws + o; o += (size_t)NG * HD;
    float* hG   = ws + o; o += (size_t)NG * HD;
    float* t1   = ws + o; o += (size_t)NG * 128;
    float* t2   = ws + o; o += (size_t)NG * 64;
    (void)ws_size; (void)n_in; (void)in_sizes; (void)out_size;

    const int EB = (N_EDGES + 7) / 8;   // wave-per-edge blocks
    const int NB = (N_NODES + 7) / 8;

    // ---- input projection: x1 = lrelu(x0 @ lin1_w^T + b) ----
    gemm(stream, x0, nullptr, nullptr, IN_DIM, lin1_w, lin1_b, x1,
         N_NODES, HD, IN_DIM, ACT_LRELU);

    // ---- GATEConv ----
    // xj = lrelu(concat(x1[src], edge_attr) @ g_lin1_w^T)
    gemm(stream, x1, src, edge_attr, HD, g_lin1_w, nullptr, xj,
         N_EDGES, HD, HD + EDD, ACT_LRELU);
    rowdot_kernel<<<NB, 256, 0, stream>>>(x1, g_att_r, grD, N_NODES);
    edge_score_gate_kernel<<<EB, 256, 0, stream>>>(xj, g_att_l, grD, dst, eV, N_EDGES);
    fillu(stream, smax, 0u, N_NODES);
    seg_max_kernel<<<(N_EDGES + 255) / 256, 256, 0, stream>>>(eV, dst, smax, N_EDGES);
    fillf(stream, ssum, 0.f, N_NODES);
    seg_expsum_kernel<<<(N_EDGES + 255) / 256, 256, 0, stream>>>(eV, dst, smax, ssum, N_EDGES);
    gemm(stream, xj, nullptr, nullptr, HD, g_lin2_w, nullptr, mm,
         N_EDGES, HD, HD, ACT_NONE);
    fillf(stream, hb, 0.f, (long)N_NODES * HD);
    scatter_kernel<<<N_EDGES, 256, 0, stream>>>(mm, nullptr, eV, ssum, dst, hb, N_EDGES);
    bias_act_kernel<<<(unsigned)(((long)N_NODES * HD + 255) / 256), 256, 0, stream>>>(
        hb, g_bias, (long)N_NODES * HD, HD, ACT_ELU);
    // x1 = relu(gru(hb, x1))
    gemm(stream, hb, nullptr, nullptr, HD, gru0_wih, gru0_bih, gi, N_NODES, 3 * HD, HD, ACT_NONE);
    gemm(stream, x1, nullptr, nullptr, HD, gru0_whh, gru0_bhh, gh, N_NODES, 3 * HD, HD, ACT_NONE);
    gru_kernel<<<N_NODES, 256, 0, stream>>>(gi, gh, x1, x1, N_NODES, 1);

    // ---- atom-level GATv2 + GRU layers ----
    for (int l = 0; l < NLAYERS; l++) {
        const float* Wl  = atom_Wl + (size_t)l * HD * HD;
        const float* Wr  = atom_Wr + (size_t)l * HD * HD;
        const float* att = atom_att + (size_t)l * HD;
        const float* ab  = atom_bias + (size_t)l * HD;
        const float* wih = agru_wih + (size_t)l * 3 * HD * HD;
        const float* whh = agru_whh + (size_t)l * 3 * HD * HD;
        const float* bih = agru_bih + (size_t)l * 3 * HD;
        const float* bhh = agru_bhh + (size_t)l * 3 * HD;

        gemm(stream, x1, nullptr, nullptr, HD, Wl, nullptr, hl, N_NODES, HD, HD, ACT_NONE);
        gemm(stream, x1, nullptr, nullptr, HD, Wr, nullptr, hr, N_NODES, HD, HD, ACT_NONE);
        edge_score_v2_kernel<<<EB, 256, 0, stream>>>(hl, hr, src, dst, att, eV, N_EDGES);
        fillu(stream, smax, 0u, N_NODES);
        seg_max_kernel<<<(N_EDGES + 255) / 256, 256, 0, stream>>>(eV, dst, smax, N_EDGES);
        fillf(stream, ssum, 0.f, N_NODES);
        seg_expsum_kernel<<<(N_EDGES + 255) / 256, 256, 0, stream>>>(eV, dst, smax, ssum, N_EDGES);
        fillf(stream, hb, 0.f, (long)N_NODES * HD);
        scatter_kernel<<<N_EDGES, 256, 0, stream>>>(hl, src, eV, ssum, dst, hb, N_EDGES);
        bias_act_kernel<<<(unsigned)(((long)N_NODES * HD + 255) / 256), 256, 0, stream>>>(
            hb, ab, (long)N_NODES * HD, HD, ACT_ELU);
        gemm(stream, hb, nullptr, nullptr, HD, wih, bih, gi, N_NODES, 3 * HD, HD, ACT_NONE);
        gemm(stream, x1, nullptr, nullptr, HD, whh, bhh, gh, N_NODES, 3 * HD, HD, ACT_NONE);
        gru_kernel<<<N_NODES, 256, 0, stream>>>(gi, gh, x1, x1, N_NODES, 1);
    }

    // ---- molecule-level readout ----
    fillf(stream, outG, 0.f, (long)NG * HD);
    scatter_kernel<<<N_NODES, 256, 0, stream>>>(x1, nullptr, nullptr, nullptr, batch, outG, N_NODES);
    bias_act_kernel<<<(unsigned)(((long)NG * HD + 255) / 256), 256, 0, stream>>>(
        outG, nullptr, (long)NG * HD, HD, ACT_RELU);
    // hl = x1 @ mol_Wl^T (constant across timesteps)
    gemm(stream, x1, nullptr, nullptr, HD, mol_Wl, nullptr, hl, N_NODES, HD, HD, ACT_NONE);
    for (int t = 0; t < NSTEPS; t++) {
        gemm(stream, outG, nullptr, nullptr, HD, mol_Wr, nullptr, hr, NG, HD, HD, ACT_NONE);
        edge_score_v2_kernel<<<NB, 256, 0, stream>>>(hl, hr, nullptr, batch, mol_att, eV, N_NODES);
        fillu(stream, smax, 0u, NG);
        seg_max_kernel<<<(N_NODES + 255) / 256, 256, 0, stream>>>(eV, batch, smax, N_NODES);
        fillf(stream, ssum, 0.f, NG);
        seg_expsum_kernel<<<(N_NODES + 255) / 256, 256, 0, stream>>>(eV, batch, smax, ssum, N_NODES);
        fillf(stream, hG, 0.f, (long)NG * HD);
        scatter_kernel<<<N_NODES, 256, 0, stream>>>(hl, nullptr, eV, ssum, batch, hG, N_NODES);
        bias_act_kernel<<<(unsigned)(((long)NG * HD + 255) / 256), 256, 0, stream>>>(
            hG, mol_bias, (long)NG * HD, HD, ACT_ELU);
        gemm(stream, hG, nullptr, nullptr, HD, mgru_wih, mgru_bih, gi, NG, 3 * HD, HD, ACT_NONE);
        gemm(stream, outG, nullptr, nullptr, HD, mgru_whh, mgru_bhh, gh, NG, 3 * HD, HD, ACT_NONE);
        gru_kernel<<<NG, 256, 0, stream>>>(gi, gh, outG, outG, NG, 1);
    }

    // ---- MLP head ----
    gemm(stream, outG, nullptr, nullptr, HD, mlp_w1, mlp_b1, t1, NG, 128, HD, ACT_RELU);
    gemm(stream, t1, nullptr, nullptr, 128, mlp_w2, mlp_b2, t2, NG, 64, 128, ACT_RELU);
    gemm(stream, t2, nullptr, nullptr, 64, mlp_w3, mlp_b3, (float*)d_out, NG, 1, 64, ACT_NONE);
}
